// BiLSTM_CRF_20091857011545
// MI455X (gfx1250) — matrix operand order repaired and checked by reading, not compiled, not run on hardware
//
#include <hip/hip_runtime.h>
#include <hip/hip_bf16.h>
#include <stdint.h>

typedef __attribute__((ext_vector_type(16))) __bf16 v16bf;
typedef __attribute__((ext_vector_type(8)))  float  v8f;

#define T_LEN 1024
#define BATCH 64
#define EDIM  128
#define HDIM  128
#define G4H   512
#define KTAG  9
#define NEGV  -10000.0f

// ---------- WMMA fragment helpers (contiguous K = 16*half + j, same order for A and B) ----------

// A-matrix 16x32 bf16 fragment. Caller passes pointer to row*stride + kbase,
// where kbase = k0 + 16*(lane>=16). [0..15] -> K=kbase+0..15.
__device__ inline v16bf afrag_bf16(const __bf16* p) {
  v16bf r;
#pragma unroll
  for (int j = 0; j < 16; ++j) r[j] = p[j];
  return r;
}

// B-matrix 32x16 bf16 fragment from row-major W[n][k]:
// lane col n, 16 contiguous K starting at k0 + 16*(lane>=16).
__device__ inline v16bf bfrag_bf16(const __bf16* p) {
  v16bf r;
#pragma unroll
  for (int j = 0; j < 16; ++j) r[j] = p[j];
  return r;
}

__device__ inline v8f wmma_bf16(v16bf a, v16bf b, v8f c) {
  return __builtin_amdgcn_wmma_f32_16x16x32_bf16(false, a, false, b,
                                                 (short)0, c, false, false);
}

__device__ inline float sigmoidf_(float x) { return 1.0f / (1.0f + expf(-x)); }

// ---------- Kernel W: one-shot f32 -> bf16 weight conversion ----------
__global__ void f32_to_bf16_kernel(const float* __restrict__ src,
                                   __bf16* __restrict__ dst, int n) {
  int i = blockIdx.x * blockDim.x + threadIdx.x;
  if (i < n) dst[i] = (__bf16)src[i];
}

// ---------- Kernel 0: embedding gather -> bf16 rows [T*B][E] ----------
__global__ void gather_embed_kernel(const int* __restrict__ sent,
                                    const float* __restrict__ embed,
                                    __bf16* __restrict__ xbf) {
  size_t idx = (size_t)blockIdx.x * blockDim.x + threadIdx.x;  // one element
  int m   = (int)(idx >> 7);      // row in [T*B)
  int col = (int)(idx & 127);
  int t = m >> 6, b = m & 63;
  int tok = sent[(size_t)b * T_LEN + t];
  xbf[idx] = (__bf16)embed[(size_t)tok * EDIM + col];
}

// ---------- Kernel 1: input gates GEMM: gates[d][m][n] = x@W_ih^T + b ----------
// One wave per 16x16 tile; wid = dir*131072 + mTile*32 + nTile.
__global__ void gates_kernel(const __bf16* __restrict__ xbf,
                             const __bf16* __restrict__ wih_f_bf,
                             const __bf16* __restrict__ wih_b_bf,
                             const float* __restrict__ b_ih_f,
                             const float* __restrict__ b_hh_f,
                             const float* __restrict__ b_ih_b,
                             const float* __restrict__ b_hh_b,
                             float* __restrict__ gates_f,
                             float* __restrict__ gates_b) {
  int wid  = blockIdx.x * 8 + (threadIdx.x >> 5);
  int lane = threadIdx.x & 31;
  int dir   = wid >> 17;
  int mTile = (wid >> 5) & 4095;
  int nTile = wid & 31;
  const __bf16* w  = dir ? wih_b_bf : wih_f_bf;
  const float* bi = dir ? b_ih_b : b_ih_f;
  const float* bh = dir ? b_hh_b : b_hh_f;
  float* out = dir ? gates_b : gates_f;

  int m0 = mTile * 16, n0 = nTile * 16;
  int col = lane & 15, half = lane >> 4;

  float bias = bi[n0 + col] + bh[n0 + col];
  v8f acc;
#pragma unroll
  for (int v = 0; v < 8; ++v) acc[v] = bias;

#pragma unroll
  for (int k0 = 0; k0 < EDIM; k0 += 32) {
    const __bf16* pa = xbf + (size_t)(m0 + col) * EDIM + k0 + half * 16;
    v16bf a = afrag_bf16(pa);
    const __bf16* pb = w + (size_t)(n0 + col) * EDIM + k0 + half * 16;
    v16bf b = bfrag_bf16(pb);
    acc = wmma_bf16(a, b, acc);
  }
#pragma unroll
  for (int v = 0; v < 8; ++v) {
    int row = m0 + v + half * 8;
    out[(size_t)row * G4H + n0 + col] = acc[v];
  }
}

// ---------- Kernel 2: sequential LSTM recurrence (1 WGP per direction) ----------
// blockIdx.x: 0 = forward, 1 = backward. 1024 threads = 32 waves.
// Wave (mTile = w>>3, j = w&7) owns the 4 gate tiles (i,f,g,o) of units
// [16j,16j+16) for batch rows [16*mTile, 16*mTile+16): elementwise LSTM update
// is fully in-register. W_hh (bf16, 128 KB) is async-DMA'd into LDS
// (global_load_async_to_lds_b128, ASYNCcnt) -- only possible on CDNA5's
// 320 KB/WGP LDS.
__global__ void __launch_bounds__(1024, 1)
lstm_kernel(const float* __restrict__ gates_in_f,
            const float* __restrict__ gates_in_b,
            const __bf16* __restrict__ whh_f_bf,
            const __bf16* __restrict__ whh_b_bf,
            __bf16* __restrict__ h_f, __bf16* __restrict__ h_b) {
  __shared__ __bf16 whh[G4H * HDIM];   // 131072 B
  __shared__ __bf16 hld[BATCH * HDIM]; // 16384 B

  int dir = blockIdx.x;
  const float* gin  = dir ? gates_in_b : gates_in_f;
  const __bf16* wsrc = dir ? whh_b_bf : whh_f_bf;
  __bf16* hout = dir ? h_b : h_f;

  int tid = threadIdx.x;

  // Async copy of W_hh (bf16) global -> LDS: 8192 x b128, 8 per thread.
  {
    unsigned ldsbase = (unsigned)(uintptr_t)(&whh[0]);
#pragma unroll
    for (int r = 0; r < 8; ++r) {
      unsigned byteoff = (unsigned)(tid * 16 + r * 1024 * 16);
      unsigned dst = ldsbase + byteoff;
      asm volatile("global_load_async_to_lds_b128 %0, %1, %2"
                   :: "v"(dst), "v"(byteoff), "s"(wsrc)
                   : "memory");
    }
  }
  for (int i = tid; i < BATCH * HDIM; i += 1024) hld[i] = (__bf16)0.0f;
  asm volatile("s_wait_asynccnt 0" ::: "memory");
  __syncthreads();

  int wave = tid >> 5, lane = tid & 31;
  int mTile = wave >> 3, j = wave & 7;
  int col = lane & 15, half = lane >> 4;
  int unit = j * 16 + col;

  float c[8];
#pragma unroll
  for (int v = 0; v < 8; ++v) c[v] = 0.0f;

  for (int t = 0; t < T_LEN; ++t) {
    int tt = dir ? (T_LEN - 1 - t) : t;
    int tn = dir ? (tt - 1) : (tt + 1);
    if (tn >= 0 && tn < T_LEN)
      __builtin_prefetch(gin + ((size_t)tn * BATCH + mTile * 16) * G4H, 0, 1);

    // A fragments: previous h (bf16) from LDS.
    v16bf a[4];
#pragma unroll
    for (int kk = 0; kk < 4; ++kk) {
      const __bf16* pa = hld + (size_t)(mTile * 16 + col) * HDIM + kk * 32 + half * 16;
      a[kk] = afrag_bf16(pa);
    }

    // 4 gate tiles: acc[g] = h @ W_hh[g]^T for this wave's unit block.
    v8f acc[4];
#pragma unroll
    for (int g = 0; g < 4; ++g) {
      v8f z;
#pragma unroll
      for (int v = 0; v < 8; ++v) z[v] = 0.0f;
#pragma unroll
      for (int kk = 0; kk < 4; ++kk) {
        const __bf16* pb = whh + (size_t)(g * HDIM + unit) * HDIM + kk * 32 + half * 16;
        v16bf b = bfrag_bf16(pb);
        z = wmma_bf16(a[kk], b, z);
      }
      acc[g] = z;
    }

    // In-register LSTM cell update (c stays f32 in VGPRs).
    float hv[8];
#pragma unroll
    for (int v = 0; v < 8; ++v) {
      int brow = mTile * 16 + v + half * 8;
      const float* gp = gin + ((size_t)tt * BATCH + brow) * G4H;
      float ig = acc[0][v] + gp[unit];
      float fg = acc[1][v] + gp[HDIM + unit];
      float gg = acc[2][v] + gp[2 * HDIM + unit];
      float og = acc[3][v] + gp[3 * HDIM + unit];
      float is = sigmoidf_(ig), fs = sigmoidf_(fg), os = sigmoidf_(og);
      float gt = tanhf(gg);
      c[v] = fs * c[v] + is * gt;
      hv[v] = os * tanhf(c[v]);
    }

    __syncthreads();  // all waves done reading old h
#pragma unroll
    for (int v = 0; v < 8; ++v) {
      int brow = mTile * 16 + v + half * 8;
      __bf16 hb = (__bf16)hv[v];
      hld[(size_t)brow * HDIM + unit] = hb;
      hout[((size_t)tt * BATCH + brow) * HDIM + unit] = hb;
    }
    __syncthreads();  // new h visible for next step
  }
}

// ---------- Kernel 3: output projection feats = [h_f|h_b] @ W_out^T + b ----------
// N padded 9 -> 16; one wave per 16-row tile, K = 256 (8 chunks).
__global__ void feats_kernel(const __bf16* __restrict__ h_f,
                             const __bf16* __restrict__ h_b,
                             const __bf16* __restrict__ wout_bf,
                             const float* __restrict__ b_out,
                             float* __restrict__ feats) {
  int wid  = blockIdx.x * 8 + (threadIdx.x >> 5);
  int lane = threadIdx.x & 31;
  int m0 = wid * 16;
  int col = lane & 15, half = lane >> 4;

  v8f acc;
  float bias = (col < KTAG) ? b_out[col] : 0.0f;
#pragma unroll
  for (int v = 0; v < 8; ++v) acc[v] = bias;

#pragma unroll
  for (int kc = 0; kc < 8; ++kc) {
    int k0 = kc * 32;
    const __bf16* hsrc = (k0 < HDIM) ? h_f : h_b;
    int kloc = k0 & (HDIM - 1);
    const __bf16* pa = hsrc + (size_t)(m0 + col) * HDIM + kloc + half * 16;
    v16bf a = afrag_bf16(pa);
    v16bf b;
    if (col < KTAG) {
      const __bf16* pb = wout_bf + (size_t)col * (2 * HDIM) + k0 + half * 16;
#pragma unroll
      for (int jj = 0; jj < 16; ++jj) b[jj] = pb[jj];
    } else {
#pragma unroll
      for (int jj = 0; jj < 16; ++jj) b[jj] = (__bf16)0.0f;
    }
    acc = wmma_bf16(a, b, acc);
  }
#pragma unroll
  for (int v = 0; v < 8; ++v)
    feats[(size_t)(m0 + v + half * 8) * 16 + col] = acc[v];
}

// ---------- Kernel 4: Viterbi (forward trellis + backtrace), 1 block ----------
__global__ void viterbi_kernel(const float* __restrict__ feats,
                               const float* __restrict__ trans,
                               unsigned char* __restrict__ bp,
                               float* __restrict__ out) {
  __shared__ float delta[2][BATCH][KTAG];
  __shared__ int bestlast[BATCH];
  int tid = threadIdx.x;               // 576 = 64*9 threads
  int b = tid / KTAG, nxt = tid % KTAG;

  float tr[KTAG];
#pragma unroll
  for (int p = 0; p < KTAG; ++p) tr[p] = trans[nxt * KTAG + p];

  delta[0][b][nxt] = (nxt == 7) ? 0.0f : NEGV;   // START = 7
  __syncthreads();

  int cur = 0;
  for (int t = 0; t < T_LEN; ++t) {
    float best = -3.0e38f; int arg = 0;
#pragma unroll
    for (int p = 0; p < KTAG; ++p) {
      float s = delta[cur][b][p] + tr[p];
      if (s > best) { best = s; arg = p; }
    }
    bp[((size_t)t * BATCH + b) * 16 + nxt] = (unsigned char)arg;
    delta[cur ^ 1][b][nxt] = best + feats[((size_t)t * BATCH + b) * 16 + nxt];
    __syncthreads();
    cur ^= 1;
  }

  if (nxt == 0) {
    float best = -3.0e38f; int arg = 0;
#pragma unroll
    for (int k = 0; k < KTAG; ++k) {
      float s = delta[cur][b][k] + trans[8 * KTAG + k];  // STOP = 8
      if (s > best) { best = s; arg = k; }
    }
    out[b] = best;
    bestlast[b] = arg;
  }
  __syncthreads();

  if (tid < BATCH) {
    int bb = tid;
    int tag = bestlast[bb];
    out[BATCH + (size_t)bb * T_LEN + (T_LEN - 1)] = (float)tag;
    for (int t = T_LEN - 1; t >= 1; --t) {
      tag = bp[((size_t)t * BATCH + bb) * 16 + tag];
      out[BATCH + (size_t)bb * T_LEN + (t - 1)] = (float)tag;
    }
  }
}

// ---------- Launch ----------
extern "C" void kernel_launch(void* const* d_in, const int* in_sizes, int n_in,
                              void* d_out, int out_size, void* d_ws, size_t ws_size,
                              hipStream_t stream) {
  const int*   sentence = (const int*)d_in[0];
  const float* embed    = (const float*)d_in[1];
  const float* w_ih_f   = (const float*)d_in[2];
  const float* w_hh_f   = (const float*)d_in[3];
  const float* b_ih_f   = (const float*)d_in[4];
  const float* b_hh_f   = (const float*)d_in[5];
  const float* w_ih_b   = (const float*)d_in[6];
  const float* w_hh_b   = (const float*)d_in[7];
  const float* b_ih_b   = (const float*)d_in[8];
  const float* b_hh_b   = (const float*)d_in[9];
  const float* w_out    = (const float*)d_in[10];
  const float* b_out    = (const float*)d_in[11];
  const float* trans    = (const float*)d_in[12];

  char* ws = (char*)d_ws;
  const size_t MROWS = (size_t)T_LEN * BATCH;          // 65536
  const size_t WN = (size_t)G4H * EDIM;                // 65536 weights per matrix
  size_t off = 0;
  __bf16* xbf    = (__bf16*)(ws + off); off += MROWS * EDIM * sizeof(__bf16);   // 16 MB
  float*  gatesF = (float*)(ws + off);  off += MROWS * G4H * sizeof(float);     // 128 MB
  float*  gatesB = (float*)(ws + off);  off += MROWS * G4H * sizeof(float);     // 128 MB
  __bf16* hF     = (__bf16*)(ws + off); off += MROWS * HDIM * sizeof(__bf16);   // 16 MB
  __bf16* hB     = (__bf16*)(ws + off); off += MROWS * HDIM * sizeof(__bf16);   // 16 MB
  float*  feats  = (float*)(ws + off);  off += MROWS * 16 * sizeof(float);      // 4 MB
  unsigned char* bp = (unsigned char*)(ws + off); off += MROWS * 16;            // 1 MB
  __bf16* wihF   = (__bf16*)(ws + off); off += WN * sizeof(__bf16);             // 128 KB
  __bf16* wihB   = (__bf16*)(ws + off); off += WN * sizeof(__bf16);
  __bf16* whhF   = (__bf16*)(ws + off); off += WN * sizeof(__bf16);
  __bf16* whhB   = (__bf16*)(ws + off); off += WN * sizeof(__bf16);
  __bf16* woutB  = (__bf16*)(ws + off); off += 4096 * sizeof(__bf16);

  // W) one-shot weight conversion to bf16
  f32_to_bf16_kernel<<<256, 256, 0, stream>>>(w_ih_f, wihF, (int)WN);
  f32_to_bf16_kernel<<<256, 256, 0, stream>>>(w_ih_b, wihB, (int)WN);
  f32_to_bf16_kernel<<<256, 256, 0, stream>>>(w_hh_f, whhF, (int)WN);
  f32_to_bf16_kernel<<<256, 256, 0, stream>>>(w_hh_b, whhB, (int)WN);
  f32_to_bf16_kernel<<<9,   256, 0, stream>>>(w_out,  woutB, KTAG * 2 * HDIM);

  // 0) embedding gather -> bf16
  gather_embed_kernel<<<32768, 256, 0, stream>>>(sentence, embed, xbf);
  // 1) input gate GEMMs (both directions): 262144 wave-tiles
  gates_kernel<<<32768, 256, 0, stream>>>(xbf, wihF, wihB,
                                          b_ih_f, b_hh_f, b_ih_b, b_hh_b,
                                          gatesF, gatesB);
  // 2) sequential recurrence, one WGP per direction
  lstm_kernel<<<2, 1024, 0, stream>>>(gatesF, gatesB, whhF, whhB, hF, hB);
  // 3) output projection
  feats_kernel<<<512, 256, 0, stream>>>(hF, hB, woutB, b_out, feats);
  // 4) Viterbi decode
  viterbi_kernel<<<1, 576, 0, stream>>>(feats, trans, bp, (float*)d_out);
}
